// BlockForNormalWindow_36155034697733
// MI455X (gfx1250) — matrix operand order repaired and checked by reading, not compiled, not run on hardware
//
#include <hip/hip_runtime.h>
#include <cmath>

// ---------------------------------------------------------------------------
// CDNA5 (gfx1250) implementation of a SAM-style windowed-attention block.
// All GEMMs run on v_wmma_f32_16x16x32_f16 (f16 in, f32 accumulate).
// Tile staging uses GLOBAL_LOAD_ASYNC_TO_LDS_B128 (ASYNCcnt-tracked) when the
// toolchain exposes the builtin, otherwise a manual LDS copy.
// ---------------------------------------------------------------------------

typedef _Float16 f16_t;
typedef __attribute__((ext_vector_type(16))) _Float16 v16h;
typedef __attribute__((ext_vector_type(8)))  float    v8f;
typedef __attribute__((ext_vector_type(4)))  int      v4i;

#define WMMA_F16(a, b, c) \
  __builtin_amdgcn_wmma_f32_16x16x32_f16(false, (a), false, (b), (short)0, (c), false, false)

#if defined(__has_builtin)
#  if __has_builtin(__builtin_amdgcn_global_load_async_to_lds_b128)
#    define HAVE_ASYNC_LDS 1
#  endif
#endif
#ifndef HAVE_ASYNC_LDS
#  define HAVE_ASYNC_LDS 0
#endif

__device__ __forceinline__ void stage16(f16_t* lds, const f16_t* g) {
#if HAVE_ASYNC_LDS
  // prototype (probe-derived): (int4 AS1* src, int4 AS3* dst, imm offset, imm cpol)
  __builtin_amdgcn_global_load_async_to_lds_b128(
      (__attribute__((address_space(1))) v4i*)(f16_t*)g,
      (__attribute__((address_space(3))) v4i*)lds, 0, 0);
#else
  *(uint4*)lds = *(const uint4*)g;
#endif
}
__device__ __forceinline__ void stage_wait() {
#if HAVE_ASYNC_LDS
  asm volatile("s_wait_asynccnt 0x0" ::: "memory");
#endif
}

// ----- problem constants ----------------------------------------------------
constexpr int Bb   = 8;
constexpr int Hh   = 64;
constexpr int Ww   = 64;
constexpr int DIM  = 384;
constexpr int NH   = 6;
constexpr int WS   = 14;
constexpr int HD   = 64;      // DIM / NH
constexpr int MLP  = 1536;
constexpr int NTOK = WS * WS; // 196 tokens per window
constexpr int ROWS = 224;     // padded tokens per window: 7*32 (WMMA K granule)
constexpr int NWIN = Bb * 5 * 5;     // 200 windows (70/14 = 5 per dim)
constexpr int HW   = NWIN * NH;      // 1200 head-windows
constexpr int MQKV = NWIN * ROWS;    // 44800 GEMM rows (windowed)
constexpr int MTOK = Bb * Hh * Ww;   // 32768 tokens (MLP rows)
constexpr float SCALE = 0.125f;      // HD^-0.5
constexpr float NEG_BIG = -1e30f;

// ----- workspace layout (bytes); buffers reused across stages ---------------
constexpr size_t OFF_WQKV = 0;
constexpr size_t OFF_WPROJ = OFF_WQKV + (size_t)DIM * 3 * DIM * 2;
constexpr size_t OFF_WFC1  = OFF_WPROJ + (size_t)DIM * DIM * 2;
constexpr size_t OFF_WFC2  = OFF_WFC1 + (size_t)DIM * MLP * 2;
constexpr size_t OFF_HWIN  = OFF_WFC2 + (size_t)MLP * DIM * 2;  // [MQKV][DIM] f16
constexpr size_t OFF_Q     = OFF_HWIN + (size_t)MQKV * DIM * 2; // [HW][ROWS][HD] f16
constexpr size_t OFF_K     = OFF_Q + (size_t)HW * ROWS * HD * 2;
constexpr size_t OFF_V     = OFF_K + (size_t)HW * ROWS * HD * 2;
constexpr size_t OFF_AO    = OFF_V + (size_t)HW * ROWS * HD * 2; // [MQKV][DIM] f16
constexpr size_t OFF_H2    = OFF_HWIN; // reuse (hwin dead after qkv GEMM)
constexpr size_t OFF_MLP1  = OFF_Q;    // reuse (q/k/v dead after attention)
constexpr size_t WS_NEEDED = OFF_AO + (size_t)MQKV * DIM * 2;   // ~176 MB

// ---------------------------------------------------------------------------
// weight f32 -> f16 conversion
// ---------------------------------------------------------------------------
__global__ __launch_bounds__(256)
void cvt_f16_kernel(const float* __restrict__ in, f16_t* __restrict__ out, int n) {
  int i = blockIdx.x * 256 + threadIdx.x;
  if (i < n) out[i] = (f16_t)in[i];
}

// ---------------------------------------------------------------------------
// LayerNorm1 + window partition (pad to 70x70, zeros outside; 224-row pad)
// one block = one token row, 384 threads (12 waves)
// ---------------------------------------------------------------------------
__global__ __launch_bounds__(384)
void ln1_window_kernel(const float* __restrict__ x, const float* __restrict__ g,
                       const float* __restrict__ b, f16_t* __restrict__ out) {
  int row = blockIdx.x;          // [0, MQKV)
  int d   = threadIdx.x;         // [0, 384)
  f16_t* dst = out + (size_t)row * DIM;
  int w = row / ROWS, rr = row - w * ROWS;
  bool valid = false;
  size_t src = 0;
  if (rr < NTOK) {
    int bb = w / 25, wy = (w % 25) / 5, wx = w % 5;
    int gy = wy * WS + rr / WS, gx = wx * WS + rr % WS;
    if (gy < Hh && gx < Ww) {
      valid = true;
      src = ((((size_t)bb * Hh + gy) * Ww) + gx) * DIM;
    }
  }
  if (!valid) { dst[d] = (f16_t)0.f; return; }   // uniform per block
  float v = x[src + d];
  float s = v, q = v * v;
  #pragma unroll
  for (int off = 16; off > 0; off >>= 1) {
    s += __shfl_xor(s, off, 32);
    q += __shfl_xor(q, off, 32);
  }
  __shared__ float wsum[12], wsq[12];
  int wv = d >> 5, lane = d & 31;
  if (lane == 0) { wsum[wv] = s; wsq[wv] = q; }
  __syncthreads();
  float ts = 0.f, tq = 0.f;
  #pragma unroll
  for (int i = 0; i < 12; ++i) { ts += wsum[i]; tq += wsq[i]; }
  float mean = ts * (1.f / DIM);
  float var  = tq * (1.f / DIM) - mean * mean;
  float inv  = rsqrtf(var + 1e-5f);
  dst[d] = (f16_t)((v - mean) * inv * g[d] + b[d]);
}

// ---------------------------------------------------------------------------
// LayerNorm2 (plain token-major)
// ---------------------------------------------------------------------------
__global__ __launch_bounds__(384)
void ln2_kernel(const float* __restrict__ x, const float* __restrict__ g,
                const float* __restrict__ b, f16_t* __restrict__ out) {
  int row = blockIdx.x;
  int d   = threadIdx.x;
  float v = x[(size_t)row * DIM + d];
  float s = v, q = v * v;
  #pragma unroll
  for (int off = 16; off > 0; off >>= 1) {
    s += __shfl_xor(s, off, 32);
    q += __shfl_xor(q, off, 32);
  }
  __shared__ float wsum[12], wsq[12];
  int wv = d >> 5, lane = d & 31;
  if (lane == 0) { wsum[wv] = s; wsq[wv] = q; }
  __syncthreads();
  float ts = 0.f, tq = 0.f;
  #pragma unroll
  for (int i = 0; i < 12; ++i) { ts += wsum[i]; tq += wsq[i]; }
  float mean = ts * (1.f / DIM);
  float var  = tq * (1.f / DIM) - mean * mean;
  float inv  = rsqrtf(var + 1e-5f);
  out[(size_t)row * DIM + d] = (f16_t)((v - mean) * inv * g[d] + b[d]);
}

// ---------------------------------------------------------------------------
// Generic tiled WMMA GEMM: C[M,N] = A[M,K] @ B[K,N]  (A,B f16 row-major)
// block tile 128x64, 256 threads = 8 waves; each wave computes a 2x2 grid of
// 16x16 tiles (4 WMMAs per k-step from 2 A + 2 B fragments). A staged [m][k]
// (async global->LDS), B staged transposed [n][k] so fragment loads are
// contiguous 16B chunks matching the wave32 WMMA layouts.
// NOTE: 16-row tiles never cross a 224-row window boundary (224 % 16 == 0),
// so window decomposition is hoisted out of the per-row epilogue loops.
// ---------------------------------------------------------------------------
enum { MODE_QKV = 0, MODE_PROJ = 1, MODE_FC1 = 2, MODE_FC2 = 3 };

template <int MODE>
__global__ __launch_bounds__(256)
void gemm_kernel(const f16_t* __restrict__ A, const f16_t* __restrict__ Bm,
                 const float* __restrict__ bias, int M, int N, int K,
                 f16_t* __restrict__ o16a, f16_t* __restrict__ o16b,
                 f16_t* __restrict__ o16c, const float* __restrict__ addin,
                 float* __restrict__ o32) {
  __shared__ f16_t sA[128 * 32];  // [m][k]
  __shared__ f16_t sBT[64 * 32];  // [n][k] (transposed at fill)
  int tid = threadIdx.x, wv = tid >> 5, lane = tid & 31;
  int half_ = lane >> 4, r = lane & 15;
  int bm = blockIdx.y * 128, bn = blockIdx.x * 64;
  int mtBase = (wv >> 1) * 2;            // {0,2,4,6}
  int ntBase = (wv & 1) * 2;             // {0,2}
  int brow = tid >> 3, bpart = tid & 7;  // B: 32 rows x 8 uint4
  v8f acc00 = {}, acc01 = {}, acc10 = {}, acc11 = {};

  for (int k0 = 0; k0 < K; k0 += 32) {
    if (k0) __syncthreads();
    // stage A tile (straight copy -> async-to-LDS)
    #pragma unroll
    for (int jj = 0; jj < 2; ++jj) {
      int i = tid + 256 * jj;
      int arow = i >> 2, apart = i & 3;
      stage16(&sA[arow * 32 + apart * 8],
              A + (size_t)(bm + arow) * K + k0 + apart * 8);
    }
    // stage B tile transposed (manual)
    {
      f16_t tmp[8];
      *(uint4*)tmp = *(const uint4*)(Bm + (size_t)(k0 + brow) * N + bn + bpart * 8);
      #pragma unroll
      for (int j = 0; j < 8; ++j) sBT[(bpart * 8 + j) * 32 + brow] = tmp[j];
    }
    stage_wait();
    __syncthreads();
    // A fragments: elem e -> K = 16*half_ + e
    const f16_t* ap0 = &sA[(mtBase * 16 + r) * 32];
    const f16_t* ap1 = ap0 + 16 * 32;
    v16h a0, a1;
    ((uint4*)&a0)[0] = *(const uint4*)(ap0 + 16 * half_);
    ((uint4*)&a0)[1] = *(const uint4*)(ap0 + 16 * half_ + 8);
    ((uint4*)&a1)[0] = *(const uint4*)(ap1 + 16 * half_);
    ((uint4*)&a1)[1] = *(const uint4*)(ap1 + 16 * half_ + 8);
    // B fragments: elem e -> K = 16*half_ + e (contiguous in sBT row)
    const f16_t* bp0 = &sBT[(ntBase * 16 + r) * 32 + 16 * half_];
    const f16_t* bp1 = bp0 + 16 * 32;
    v16h b0, b1;
    ((uint4*)&b0)[0] = *(const uint4*)(bp0);
    ((uint4*)&b0)[1] = *(const uint4*)(bp0 + 8);
    ((uint4*)&b1)[0] = *(const uint4*)(bp1);
    ((uint4*)&b1)[1] = *(const uint4*)(bp1 + 8);
    acc00 = WMMA_F16(a0, b0, acc00);
    acc01 = WMMA_F16(a0, b1, acc01);
    acc10 = WMMA_F16(a1, b0, acc10);
    acc11 = WMMA_F16(a1, b1, acc11);
  }

  v8f accs[2][2] = {{acc00, acc01}, {acc10, acc11}};
  #pragma unroll
  for (int im = 0; im < 2; ++im) {
    int rowbase = bm + (mtBase + im) * 16;   // tile-uniform
    int w  = rowbase / ROWS;                 // hoisted division
    int rr0 = rowbase - w * ROWS + 8 * half_;
    #pragma unroll
    for (int jn = 0; jn < 2; ++jn) {
      int gcol = bn + (ntBase + jn) * 16 + r;
      float bv = bias[gcol];
      if constexpr (MODE == MODE_QKV) {
        int which = gcol / DIM, cc = gcol - which * DIM;
        int head = cc >> 6, col = cc & 63;
        f16_t* dst = (which == 0) ? o16a : (which == 1) ? o16b : o16c;
        size_t dbase = (((size_t)(w * NH + head)) * ROWS + rr0) * HD + col;
        #pragma unroll
        for (int vr = 0; vr < 8; ++vr)
          dst[dbase + (size_t)vr * HD] = (f16_t)(accs[im][jn][vr] + bv);
      } else if constexpr (MODE == MODE_PROJ) {
        int bb = w / 25, wy = (w % 25) / 5, wx = w % 5;
        #pragma unroll
        for (int vr = 0; vr < 8; ++vr) {
          int rr = rr0 + vr;
          float val = accs[im][jn][vr] + bv;
          if (rr < NTOK) {
            int gy = wy * WS + rr / WS, gx = wx * WS + rr % WS;
            if (gy < Hh && gx < Ww) {
              size_t idx = ((((size_t)bb * Hh + gy) * Ww) + gx) * DIM + gcol;
              o32[idx] = val + addin[idx];   // residual-1 + window unpartition
            }
          }
        }
      } else if constexpr (MODE == MODE_FC1) {
        size_t obase = (size_t)(rowbase + 8 * half_) * N + gcol;
        #pragma unroll
        for (int vr = 0; vr < 8; ++vr) {
          float val = accs[im][jn][vr] + bv;
          float ge = 0.5f * val * (1.f + erff(val * 0.70710678118654752f));
          o16a[obase + (size_t)vr * N] = (f16_t)ge;
        }
      } else { // MODE_FC2
        size_t obase = (size_t)(rowbase + 8 * half_) * DIM + gcol;
        #pragma unroll
        for (int vr = 0; vr < 8; ++vr)
          o32[obase + (size_t)vr * DIM] += accs[im][jn][vr] + bv;  // residual-2
      }
    }
  }
}

// ---------------------------------------------------------------------------
// Attention: one workgroup per head-window (1200 blocks, 8 waves).
// LDS: q[224][64]f16, k[224][64]f16, vT[64][224]f16, S[224][224]f32,
//      relH/relW[224][14]f32, rowsum[224]f32  => 312,704 B of 320 KB.
// ---------------------------------------------------------------------------
constexpr int ATT_SMEM_BYTES =
    (3 * ROWS * HD) * 2 + (ROWS * ROWS + 2 * ROWS * WS + ROWS) * 4;

__global__ __launch_bounds__(256)
void attn_kernel(const f16_t* __restrict__ q, const f16_t* __restrict__ k,
                 const f16_t* __restrict__ v, const float* __restrict__ rph,
                 const float* __restrict__ rpw, f16_t* __restrict__ out) {
  extern __shared__ __align__(16) char smem[];
  f16_t* sQ  = (f16_t*)smem;              // [ROWS][HD]
  f16_t* sK  = sQ + ROWS * HD;            // [ROWS][HD]
  f16_t* sVT = sK + ROWS * HD;            // [HD][ROWS]
  float* sS  = (float*)(sVT + HD * ROWS); // [ROWS][ROWS]
  float* sRH = sS + ROWS * ROWS;          // [ROWS][WS]
  float* sRW = sRH + ROWS * WS;           // [ROWS][WS]
  float* sSum = sRW + ROWS * WS;          // [ROWS]

  int tid = threadIdx.x, wv = tid >> 5, lane = tid & 31;
  int half_ = lane >> 4, r = lane & 15;
  int hw = blockIdx.x;
  int w = hw / NH, head = hw - w * NH;
  size_t base = (size_t)hw * ROWS * HD;

  // stage q/k (straight copy -> async) and v (manual transpose) into LDS
  for (int i = tid; i < ROWS * HD / 8; i += 256) {
    stage16(sQ + i * 8, q + base + i * 8);
    stage16(sK + i * 8, k + base + i * 8);
    int row = i >> 3, c0 = (i & 7) * 8;
    f16_t tmp[8];
    *(uint4*)tmp = ((const uint4*)(v + base))[i];
    #pragma unroll
    for (int j = 0; j < 8; ++j) sVT[(c0 + j) * ROWS + row] = tmp[j];
  }
  stage_wait();
  __syncthreads();

  // decomposed relative-position bias: relH[q][kh], relW[q][kw] (unscaled q)
  for (int i = tid; i < ROWS * WS; i += 256) {
    int m = i / WS, kk = i - (i / WS) * WS;
    float ah = 0.f, aw = 0.f;
    if (m < NTOK) {
      int hq = m / WS, wq = m - hq * WS;
      const float* ph = rph + (hq - kk + WS - 1) * HD;
      const float* pw = rpw + (wq - kk + WS - 1) * HD;
      #pragma unroll 8
      for (int c = 0; c < HD; ++c) {
        float qc = (float)sQ[m * HD + c];
        ah += qc * ph[c];
        aw += qc * pw[c];
      }
    }
    sRH[i] = ah;
    sRW[i] = aw;
  }
  __syncthreads();

  // S = SCALE * (q @ k^T) + relH + relW   (14x14 grid of 16x16 WMMA tiles)
  for (int t = wv; t < 196; t += 8) {
    int mt = t / 14, nt = t - (t / 14) * 14;
    v8f acc = {};
    #pragma unroll
    for (int ks = 0; ks < 2; ++ks) {
      const f16_t* ap = sQ + (mt * 16 + r) * HD + ks * 32;
      v16h a;
      ((uint4*)&a)[0] = *(const uint4*)(ap + 16 * half_);
      ((uint4*)&a)[1] = *(const uint4*)(ap + 16 * half_ + 8);
      const f16_t* bp = sK + (nt * 16 + r) * HD + ks * 32 + 16 * half_;
      v16h b;
      ((uint4*)&b)[0] = *(const uint4*)(bp);
      ((uint4*)&b)[1] = *(const uint4*)(bp + 8);
      acc = WMMA_F16(a, b, acc);
    }
    // additive masking: rel loads stay unconditional (no exec branches),
    // padded keys get -1e30 added -> exp() underflows to 0 in softmax.
    int key = nt * 16 + r;
    int kc = key < NTOK ? key : 0;
    int kh = kc / WS, kw = kc - kh * WS;
    float maskAdd = key < NTOK ? 0.f : NEG_BIG;
    int mrow0 = mt * 16 + 8 * half_;
    const float* pRH = sRH + mrow0 * WS + kh;
    const float* pRW = sRW + mrow0 * WS + kw;
    float* pS = sS + mrow0 * ROWS + key;
    #pragma unroll
    for (int vr = 0; vr < 8; ++vr)
      pS[vr * ROWS] = acc[vr] * SCALE + pRH[vr * WS] + pRW[vr * WS] + maskAdd;
  }
  __syncthreads();

  // row softmax (unnormalized exp; sums kept, division folded into PV epilogue)
  for (int m = wv; m < ROWS; m += 8) {
    float* row = sS + m * ROWS;
    float vals[7];
    float vmax = NEG_BIG;
    #pragma unroll
    for (int j = 0; j < 7; ++j) { vals[j] = row[lane + j * 32]; vmax = fmaxf(vmax, vals[j]); }
    #pragma unroll
    for (int off = 16; off > 0; off >>= 1) vmax = fmaxf(vmax, __shfl_xor(vmax, off, 32));
    float s = 0.f;
    #pragma unroll
    for (int j = 0; j < 7; ++j) { float e = __expf(vals[j] - vmax); vals[j] = e; s += e; }
    #pragma unroll
    for (int off = 16; off > 0; off >>= 1) s += __shfl_xor(s, off, 32);
    #pragma unroll
    for (int j = 0; j < 7; ++j) row[lane + j * 32] = vals[j];
    if (lane == 0) sSum[m] = s;
  }
  __syncthreads();

  // O = (P @ V) / rowsum ; write to windowed head-interleaved layout (f16)
  for (int t = wv; t < 56; t += 8) {
    int mt = t >> 2, nt = t & 3;
    v8f acc = {};
    for (int ks = 0; ks < 7; ++ks) {
      const float* ap = sS + (mt * 16 + r) * ROWS + ks * 32;
      v16h a;
      #pragma unroll
      for (int e = 0; e < 8; ++e) a[e] = (f16_t)ap[16 * half_ + e];
      #pragma unroll
      for (int e = 0; e < 8; ++e) a[e + 8] = (f16_t)ap[16 * half_ + 8 + e];
      const f16_t* bp = sVT + (nt * 16 + r) * ROWS + ks * 32 + 16 * half_;
      v16h b;
      ((uint4*)&b)[0] = *(const uint4*)(bp);
      ((uint4*)&b)[1] = *(const uint4*)(bp + 8);
      acc = WMMA_F16(a, b, acc);
    }
    int c = nt * 16 + r;
    int m0 = mt * 16 + 8 * half_;
    f16_t* po = out + ((size_t)w * ROWS + m0) * DIM + head * HD + c;
    const float* ps = sSum + m0;
    #pragma unroll
    for (int vr = 0; vr < 8; ++vr)
      po[(size_t)vr * DIM] = (f16_t)(acc[vr] / ps[vr]);
  }
}

// ---------------------------------------------------------------------------
extern "C" void kernel_launch(void* const* d_in, const int* in_sizes, int n_in,
                              void* d_out, int out_size, void* d_ws, size_t ws_size,
                              hipStream_t stream) {
  (void)in_sizes; (void)n_in; (void)out_size; (void)ws_size; // needs WS_NEEDED bytes
  const float* x      = (const float*)d_in[0];
  const float* ln1_w  = (const float*)d_in[1];
  const float* ln1_b  = (const float*)d_in[2];
  const float* qkv_w  = (const float*)d_in[3];
  const float* qkv_b  = (const float*)d_in[4];
  const float* proj_w = (const float*)d_in[5];
  const float* proj_b = (const float*)d_in[6];
  const float* rph    = (const float*)d_in[7];
  const float* rpw    = (const float*)d_in[8];
  const float* ln2_w  = (const float*)d_in[9];
  const float* ln2_b  = (const float*)d_in[10];
  const float* fc1_w  = (const float*)d_in[11];
  const float* fc1_b  = (const float*)d_in[12];
  const float* fc2_w  = (const float*)d_in[13];
  const float* fc2_b  = (const float*)d_in[14];
  float* out = (float*)d_out;
  char*  ws  = (char*)d_ws;

  f16_t* wqkv  = (f16_t*)(ws + OFF_WQKV);
  f16_t* wproj = (f16_t*)(ws + OFF_WPROJ);
  f16_t* wfc1  = (f16_t*)(ws + OFF_WFC1);
  f16_t* wfc2  = (f16_t*)(ws + OFF_WFC2);
  f16_t* hwin  = (f16_t*)(ws + OFF_HWIN);
  f16_t* qb    = (f16_t*)(ws + OFF_Q);
  f16_t* kb    = (f16_t*)(ws + OFF_K);
  f16_t* vb    = (f16_t*)(ws + OFF_V);
  f16_t* ao    = (f16_t*)(ws + OFF_AO);
  f16_t* h2    = (f16_t*)(ws + OFF_H2);
  f16_t* mlp1  = (f16_t*)(ws + OFF_MLP1);

  // 0) weights -> f16 (tiny; stays L2-resident afterwards)
  cvt_f16_kernel<<<(DIM * 3 * DIM + 255) / 256, 256, 0, stream>>>(qkv_w, wqkv, DIM * 3 * DIM);
  cvt_f16_kernel<<<(DIM * DIM + 255) / 256, 256, 0, stream>>>(proj_w, wproj, DIM * DIM);
  cvt_f16_kernel<<<(DIM * MLP + 255) / 256, 256, 0, stream>>>(fc1_w, wfc1, DIM * MLP);
  cvt_f16_kernel<<<(MLP * DIM + 255) / 256, 256, 0, stream>>>(fc2_w, wfc2, MLP * DIM);

  // 1) LN1 + window partition
  ln1_window_kernel<<<MQKV, 384, 0, stream>>>(x, ln1_w, ln1_b, hwin);

  // 2) QKV projection, scatter into [head-window][row][hd] f16 buffers
  gemm_kernel<MODE_QKV><<<dim3((3 * DIM) / 64, MQKV / 128), 256, 0, stream>>>(
      hwin, wqkv, qkv_b, MQKV, 3 * DIM, DIM, qb, kb, vb, nullptr, nullptr);

  // 3) attention (QK^T -> rel-pos bias -> softmax -> PV), all WMMA
  hipError_t err_ = hipFuncSetAttribute(
      (const void*)attn_kernel, hipFuncAttributeMaxDynamicSharedMemorySize,
      ATT_SMEM_BYTES);
  (void)err_;
  attn_kernel<<<HW, 256, ATT_SMEM_BYTES, stream>>>(qb, kb, vb, rph, rpw, ao);

  // 4) output projection + residual-1 + window unpartition -> d_out (f32)
  gemm_kernel<MODE_PROJ><<<dim3(DIM / 64, MQKV / 128), 256, 0, stream>>>(
      ao, wproj, proj_b, MQKV, DIM, DIM, nullptr, nullptr, nullptr, x, out);

  // 5) LN2 over x2 (d_out)
  ln2_kernel<<<MTOK, 384, 0, stream>>>(out, ln2_w, ln2_b, h2);

  // 6) fc1 + exact GELU
  gemm_kernel<MODE_FC1><<<dim3(MLP / 64, MTOK / 128), 256, 0, stream>>>(
      h2, wfc1, fc1_b, MTOK, MLP, DIM, mlp1, nullptr, nullptr, nullptr, nullptr);

  // 7) fc2 + residual-2 accumulated into d_out
  gemm_kernel<MODE_FC2><<<dim3(DIM / 64, MTOK / 128), 256, 0, stream>>>(
      mlp1, wfc2, fc2_b, MTOK, DIM, MLP, nullptr, nullptr, nullptr, nullptr, out);
}
